// RWKV_RNN_12412455485502
// MI455X (gfx1250) — compile-verified, matrix operand order checked by reading, not run on hardware
//
#include <hip/hip_runtime.h>
#include <hip/hip_bf16.h>

typedef __bf16 bf16;
typedef __attribute__((ext_vector_type(16))) __bf16 v16bf;
typedef __attribute__((ext_vector_type(8)))  __bf16 bf16x8;
typedef __attribute__((ext_vector_type(8)))  float  v8f;

#define E_DIM 2048
#define L_DIM 1024
#define B_DIM 4
#define H_DIM 32
#define S_DIM 64
#define M_DIM (B_DIM*L_DIM)   // 4096 tokens

static constexpr int BK   = 32;   // K per WMMA step (bf16)
static constexpr int LPAD = 8;    // LDS row padding (bf16 elems)

enum EpiMode {
  EPI_F32 = 0,        // Cf = acc
  EPI_TANH_BF16,      // Cb = tanh(acc)
  EPI_SIG_F32,        // Cf = sigmoid(acc)
  EPI_RELUSQ_BF16,    // Cb = relu(acc)^2
  EPI_SILU_F32,       // Cf = acc*sigmoid(acc)
  EPI_DECAY_F32,      // Cf = exp(-exp(row[n] + acc))
  EPI_GATE_BF16,      // Cb = aux0 + aux1*(row[n] + acc)
  EPI_ADD_F32,        // Cf = aux0 + acc        (residual)
  EPI_FINAL_F32       // Cf = aux0 + aux1*acc   (x1 + rr*(kk@Wfv^T))
};

// ---------------------------------------------------------------------------
// Tiled bf16 WMMA GEMM:  C[M,N] = A[M,K] * W[N,K]^T  (i.e. x @ W.T)
// 256 threads = 8 waves arranged WROWS x WCOLS; each wave does
// (BM/WROWS/16) x (BN/WCOLS/16) v_wmma_f32_16x16x32_bf16 tiles.
// Register->LDS double buffering: one barrier per K-iteration; next tile's
// global_load_b128s overlap the current tile's WMMA stream.
// ---------------------------------------------------------------------------
template<int BM, int BN, int WROWS, int WCOLS, int EPI>
__global__ __launch_bounds__(256)
void gemm_bf16_wmma(const bf16* __restrict__ A,  int lda,
                    const bf16* __restrict__ Bw, int ldb,
                    float* __restrict__ Cf, bf16* __restrict__ Cb, int ldc,
                    int K,
                    const float* __restrict__ aux0,
                    const float* __restrict__ aux1,
                    const float* __restrict__ auxrow)
{
  constexpr int TM = BM / WROWS;
  constexpr int TN = BN / WCOLS;
  constexpr int MR = TM / 16;
  constexpr int NR = TN / 16;
  constexpr int LDS_S = BK + LPAD;          // 40 bf16 rows; 16B-aligned chunks
  constexpr int THREADS = 256;
  constexpr int VEC = 8;                    // bf16x8 = 16B
  constexpr int KV = BK / VEC;              // 4 vectors per row
  constexpr int A_PASS = (BM * BK) / (THREADS * VEC);             // 2 @ BM=128
  constexpr int B_TOT  = (BN * BK) / VEC;                         // total vecs
  constexpr int B_PASS = (B_TOT + THREADS - 1) / THREADS;         // 1 or 2

  __shared__ bf16 As[2 * BM * LDS_S];
  __shared__ bf16 Bs[2 * BN * LDS_S];

  const int tid  = threadIdx.x;
  const int wave = tid >> 5;
  const int lane = tid & 31;
  const int wrow = wave / WCOLS;
  const int wcol = wave % WCOLS;
  const int lm   = lane & 15;
  const int hi   = lane >> 4;               // 0 or 1
  const int k0   = hi * 8;                  // K chunk base per ISA lane layout
  const size_t m0 = (size_t)blockIdx.y * BM;
  const size_t n0 = (size_t)blockIdx.x * BN;

  v8f acc[MR][NR];
  {
    v8f z = {};
#pragma unroll
    for (int i = 0; i < MR; ++i)
#pragma unroll
      for (int j = 0; j < NR; ++j) acc[i][j] = z;
  }

  bf16x8 aReg[A_PASS], bReg[B_PASS];

  auto load_tiles = [&](int kb) {
#pragma unroll
    for (int p = 0; p < A_PASS; ++p) {
      int vi = tid + p * THREADS;
      int r = vi / KV;
      int c = (vi % KV) * VEC;
      aReg[p] = *reinterpret_cast<const bf16x8*>(A + (m0 + (size_t)r) * lda + kb + c);
    }
#pragma unroll
    for (int p = 0; p < B_PASS; ++p) {
      int vi = tid + p * THREADS;
      if (vi < B_TOT) {
        int r = vi / KV;
        int c = (vi % KV) * VEC;
        bReg[p] = *reinterpret_cast<const bf16x8*>(Bw + (n0 + (size_t)r) * ldb + kb + c);
      }
    }
  };
  auto store_tiles = [&](int buf) {
    bf16* Ab = As + buf * (BM * LDS_S);
    bf16* Bb = Bs + buf * (BN * LDS_S);
#pragma unroll
    for (int p = 0; p < A_PASS; ++p) {
      int vi = tid + p * THREADS;
      int r = vi / KV;
      int c = (vi % KV) * VEC;
      *reinterpret_cast<bf16x8*>(&Ab[r * LDS_S + c]) = aReg[p];
    }
#pragma unroll
    for (int p = 0; p < B_PASS; ++p) {
      int vi = tid + p * THREADS;
      if (vi < B_TOT) {
        int r = vi / KV;
        int c = (vi % KV) * VEC;
        *reinterpret_cast<bf16x8*>(&Bb[r * LDS_S + c]) = bReg[p];
      }
    }
  };

  // prologue: tile 0 -> buffer 0
  load_tiles(0);
  store_tiles(0);

  const int nIter = K / BK;
  int cur = 0;
  for (int it = 0; it < nIter; ++it) {
    const int kb_next = (it + 1) * BK;
    const bool has_next = kb_next < K;
    if (has_next) {
      load_tiles(kb_next);                        // in flight during WMMAs
      if (kb_next + BK < K) {                     // warm L2 two tiles ahead
        int vi = tid;
        int r = vi / KV, c = (vi % KV) * VEC;
        __builtin_prefetch(A + (m0 + (size_t)r) * lda + kb_next + BK + c, 0, 1);
        if (vi < B_TOT)
          __builtin_prefetch(Bw + (n0 + (size_t)r) * ldb + kb_next + BK + c, 0, 1);
      }
    }
    __syncthreads();                              // buf[cur] visible to all

    const bf16* Ab = As + cur * (BM * LDS_S);
    const bf16* Bb = Bs + cur * (BN * LDS_S);
    v16bf af[MR], bfr[NR];
#pragma unroll
    for (int i = 0; i < MR; ++i) {
      const bf16* p = &Ab[(wrow * TM + i * 16 + lm) * LDS_S];
#pragma unroll
      for (int q = 0; q < 8; ++q) { af[i][q] = p[k0 + q]; af[i][8 + q] = p[k0 + 16 + q]; }
    }
#pragma unroll
    for (int j = 0; j < NR; ++j) {
      const bf16* p = &Bb[(wcol * TN + j * 16 + lm) * LDS_S];
#pragma unroll
      for (int q = 0; q < 8; ++q) { bfr[j][q] = p[k0 + q]; bfr[j][8 + q] = p[k0 + 16 + q]; }
    }
#pragma unroll
    for (int i = 0; i < MR; ++i)
#pragma unroll
      for (int j = 0; j < NR; ++j)
        acc[i][j] = __builtin_amdgcn_wmma_f32_16x16x32_bf16(
            false, af[i], false, bfr[j], (short)0, acc[i][j], false, false);

    if (has_next) store_tiles(cur ^ 1);           // other buffer: no conflict
    cur ^= 1;
  }

  // fused epilogue; C/D layout: vgpr v -> M = v + 8*hi, N = lane&15
#pragma unroll
  for (int i = 0; i < MR; ++i)
#pragma unroll
    for (int j = 0; j < NR; ++j)
#pragma unroll
      for (int v = 0; v < 8; ++v) {
        size_t m = m0 + wrow * TM + i * 16 + v + 8 * hi;
        size_t n = n0 + wcol * TN + j * 16 + lm;
        size_t idx = m * (size_t)ldc + n;
        float val = acc[i][j][v];
        if constexpr (EPI == EPI_F32)            Cf[idx] = val;
        else if constexpr (EPI == EPI_TANH_BF16) Cb[idx] = (bf16)tanhf(val);
        else if constexpr (EPI == EPI_SIG_F32)   Cf[idx] = 1.f / (1.f + expf(-val));
        else if constexpr (EPI == EPI_RELUSQ_BF16) {
          float t = fmaxf(val, 0.f); Cb[idx] = (bf16)(t * t);
        }
        else if constexpr (EPI == EPI_SILU_F32)  Cf[idx] = val / (1.f + expf(-val));
        else if constexpr (EPI == EPI_DECAY_F32) Cf[idx] = expf(-expf(auxrow[n] + val));
        else if constexpr (EPI == EPI_GATE_BF16)
          Cb[idx] = (bf16)(aux0[idx] + aux1[idx] * (auxrow[n] + val));
        else if constexpr (EPI == EPI_ADD_F32)   Cf[idx] = aux0[idx] + val;
        else if constexpr (EPI == EPI_FINAL_F32) Cf[idx] = aux0[idx] + aux1[idx] * val;
      }
}

// ---------------------------------------------------------------------------
// LayerNorm over E=2048, one 256-thread block per token
// ---------------------------------------------------------------------------
__global__ __launch_bounds__(256)
void ln_kernel(const float* __restrict__ x, const float* __restrict__ w,
               const float* __restrict__ b, float* __restrict__ y)
{
  const int token = blockIdx.x;
  const float* xr = x + (size_t)token * E_DIM;
  float* yr = y + (size_t)token * E_DIM;
  __shared__ float red[256];
  int tid = threadIdx.x;
  float s = 0.f;
  for (int e = tid; e < E_DIM; e += 256) s += xr[e];
  red[tid] = s; __syncthreads();
  for (int st = 128; st > 0; st >>= 1) { if (tid < st) red[tid] += red[tid + st]; __syncthreads(); }
  float mean = red[0] * (1.f / E_DIM);
  __syncthreads();
  float s2 = 0.f;
  for (int e = tid; e < E_DIM; e += 256) { float d = xr[e] - mean; s2 += d * d; }
  red[tid] = s2; __syncthreads();
  for (int st = 128; st > 0; st >>= 1) { if (tid < st) red[tid] += red[tid + st]; __syncthreads(); }
  float inv = rsqrtf(red[0] * (1.f / E_DIM) + 1e-5f);
  for (int e = tid; e < E_DIM; e += 256) yr[e] = (xr[e] - mean) * inv * w[e] + b[e];
}

// token shift + first mix:  sx = prev - xa ; z1 = bf16(xa + sx*maa_x)
__global__ __launch_bounds__(256)
void shift_mix1_kernel(const float* __restrict__ xa, const float* __restrict__ state,
                       const float* __restrict__ maa_x,
                       float* __restrict__ sx, bf16* __restrict__ z1)
{
  const size_t N = (size_t)M_DIM * E_DIM;
  for (size_t i = (size_t)blockIdx.x * 256 + threadIdx.x; i < N; i += (size_t)gridDim.x * 256) {
    int e = (int)(i & (E_DIM - 1));
    size_t token = i >> 11;
    int t = (int)(token & (L_DIM - 1));
    int b = (int)(token >> 10);
    float cur = xa[i];
    float prev = (t == 0) ? state[((size_t)b * (2 + S_DIM) + 1) * E_DIM + e] : xa[i - E_DIM];
    float s = prev - cur;
    sx[i] = s;
    z1[i] = (bf16)(cur + s * maa_x[e]);
  }
}

// channel-mix token shift: two gated bf16 outputs
__global__ __launch_bounds__(256)
void shift_mix2_kernel(const float* __restrict__ xc, const float* __restrict__ state,
                       const float* __restrict__ mk, const float* __restrict__ mr,
                       bf16* __restrict__ xk2, bf16* __restrict__ xr2)
{
  const size_t N = (size_t)M_DIM * E_DIM;
  for (size_t i = (size_t)blockIdx.x * 256 + threadIdx.x; i < N; i += (size_t)gridDim.x * 256) {
    int e = (int)(i & (E_DIM - 1));
    size_t token = i >> 11;
    int t = (int)(token & (L_DIM - 1));
    int b = (int)(token >> 10);
    float cur = xc[i];
    float prev = (t == 0) ? state[((size_t)b * (2 + S_DIM) + 0) * E_DIM + e] : xc[i - E_DIM];
    float s = prev - cur;
    xk2[i] = (bf16)(cur + s * mk[e]);
    xr2[i] = (bf16)(cur + s * mr[e]);
  }
}

// ---------------------------------------------------------------------------
// WKV scan: one block per (b,h). 256 threads: col j = tid&63, row-group
// ig = tid>>6 owns 16 state rows in registers. Sequential over L.
// ---------------------------------------------------------------------------
__global__ __launch_bounds__(256)
void wkv_scan_kernel(const float* __restrict__ r, const float* __restrict__ k,
                     const float* __restrict__ v, const float* __restrict__ w,
                     const float* __restrict__ u, const float* __restrict__ state,
                     float* __restrict__ out)
{
  const int bh = blockIdx.x;
  const int b = bh >> 5;
  const int h = bh & 31;
  const int tid = threadIdx.x;
  const int j = tid & 63;
  const int ig = tid >> 6;

  float s[16], ul[16];
  // state[:,2:2+S] is a contiguous [S*E] block per batch == flat [H,S,S]
  const float* sb = state + ((size_t)b * (2 + S_DIM) + 2) * E_DIM + (size_t)h * S_DIM * S_DIM;
#pragma unroll
  for (int ii = 0; ii < 16; ++ii) {
    s[ii]  = sb[(ig * 16 + ii) * S_DIM + j];
    ul[ii] = u[h * S_DIM + ig * 16 + ii];
  }

  __shared__ float rs[64], ks[64], vs[64], ws[64];
  __shared__ float red[4][64];
  const size_t hb = (size_t)h * S_DIM;

  for (int t = 0; t < L_DIM; ++t) {
    size_t row = ((size_t)b * L_DIM + t) * E_DIM + hb;
    if      (ig == 0) rs[j] = r[row + j];
    else if (ig == 1) ks[j] = k[row + j];
    else if (ig == 2) vs[j] = v[row + j];
    else              ws[j] = w[row + j];
    __syncthreads();
    float vj = vs[j];
    float p = 0.f;
#pragma unroll
    for (int ii = 0; ii < 16; ++ii) {
      int i = ig * 16 + ii;
      float a = ks[i] * vj;                       // outer product element
      p = fmaf(rs[i], fmaf(ul[ii], a, s[ii]), p); // r · (u*a + s)
      s[ii] = fmaf(ws[i], s[ii], a);              // s = a + w*s
    }
    red[ig][j] = p;
    __syncthreads();
    if (ig == 0) out[row + j] = red[0][j] + red[1][j] + red[2][j] + red[3][j];
    __syncthreads();
  }
}

// group-norm over heads (S=64) then multiply by silu gate -> bf16
__global__ __launch_bounds__(256)
void gnorm_mul_kernel(const float* __restrict__ o, const float* __restrict__ g,
                      const float* __restrict__ gw, const float* __restrict__ gb,
                      bf16* __restrict__ ogg)
{
  const size_t base_t = (size_t)blockIdx.x * E_DIM;
  int tid = threadIdx.x;
  int h = tid >> 3, sub = tid & 7;
  int ebase = h * S_DIM + sub * 8;
  float vals[8]; float s = 0.f, s2 = 0.f;
#pragma unroll
  for (int q = 0; q < 8; ++q) { float v = o[base_t + ebase + q]; vals[q] = v; s += v; s2 += v * v; }
  __shared__ float rs[256], rq[256];
  rs[tid] = s; rq[tid] = s2; __syncthreads();
  for (int st = 4; st > 0; st >>= 1) {
    if (sub < st) { rs[tid] += rs[tid + st]; rq[tid] += rq[tid + st]; }
    __syncthreads();
  }
  float mean = rs[h * 8] * (1.f / S_DIM);
  float var  = rq[h * 8] * (1.f / S_DIM) - mean * mean;
  float inv = rsqrtf(var + 1e-5f);
#pragma unroll
  for (int q = 0; q < 8; ++q) {
    int e = ebase + q;
    float og = (vals[q] - mean) * inv * gw[e] + gb[e];
    ogg[base_t + e] = (bf16)(og * g[base_t + e]);
  }
}

// weight casts
__global__ void cast_bf16_kernel(const float* __restrict__ src, bf16* __restrict__ dst, size_t n)
{
  for (size_t i = (size_t)blockIdx.x * 256 + threadIdx.x; i < n; i += (size_t)gridDim.x * 256)
    dst[i] = (bf16)src[i];
}
__global__ void cast_tr_bf16_kernel(const float* __restrict__ src, bf16* __restrict__ dst,
                                    int R, int C)
{
  size_t n = (size_t)R * C;
  for (size_t i = (size_t)blockIdx.x * 256 + threadIdx.x; i < n; i += (size_t)gridDim.x * 256) {
    size_t r = i / C, c = i % C;
    dst[c * (size_t)R + r] = (bf16)src[i];
  }
}

// ---------------------------------------------------------------------------
extern "C" void kernel_launch(void* const* d_in, const int* in_sizes, int n_in,
                              void* d_out, int out_size, void* d_ws, size_t ws_size,
                              hipStream_t stream)
{
  const float* x          = (const float*)d_in[0];
  const float* state      = (const float*)d_in[1];
  const float* ln1_w      = (const float*)d_in[2];
  const float* ln1_b      = (const float*)d_in[3];
  const float* ln2_w      = (const float*)d_in[4];
  const float* ln2_b      = (const float*)d_in[5];
  const float* maa_x      = (const float*)d_in[6];
  const float* maa_w1     = (const float*)d_in[7];   // [E,160]
  const float* maa_w2     = (const float*)d_in[8];   // [5,32,E]
  const float* maa_stack  = (const float*)d_in[9];   // [1,5,E]
  const float* time_decay = (const float*)d_in[10];
  const float* td_w1      = (const float*)d_in[11];  // [E,64]
  const float* td_w2      = (const float*)d_in[12];  // [64,E]
  const float* u          = (const float*)d_in[13];  // [H,S,1]
  const float* Wr         = (const float*)d_in[14];
  const float* Wk         = (const float*)d_in[15];
  const float* Wv         = (const float*)d_in[16];
  const float* Wg         = (const float*)d_in[17];
  const float* Wo         = (const float*)d_in[18];
  const float* gn_w       = (const float*)d_in[19];
  const float* gn_b       = (const float*)d_in[20];
  const float* ffn_maa_k  = (const float*)d_in[21];
  const float* ffn_maa_r  = (const float*)d_in[22];
  const float* Wfk        = (const float*)d_in[23];
  const float* Wfr        = (const float*)d_in[24];
  const float* Wfv        = (const float*)d_in[25];
  (void)in_sizes; (void)n_in; (void)out_size; (void)ws_size;

  const size_t NE  = (size_t)M_DIM * E_DIM;
  const size_t EE  = (size_t)E_DIM * E_DIM;
  char* ws = (char*)d_ws;
  size_t off = 0;
  auto alloc = [&](size_t bytes) -> void* {
    void* p = ws + off;
    off = (off + bytes + 255) & ~(size_t)255;
    return p;
  };

  // fp32 activations
  float* xa  = (float*)alloc(NE * 4);
  float* sx  = (float*)alloc(NE * 4);
  float* rb  = (float*)alloc(NE * 4);
  float* kb2 = (float*)alloc(NE * 4);
  float* vb  = (float*)alloc(NE * 4);
  float* wb  = (float*)alloc(NE * 4);
  float* gb  = (float*)alloc(NE * 4);
  float* ob  = (float*)alloc(NE * 4);
  float* x1  = (float*)alloc(NE * 4);
  // bf16 activations
  bf16* z1    = (bf16*)alloc(NE * 2);
  bf16* xkb   = (bf16*)alloc(NE * 2);
  bf16* xwb   = (bf16*)alloc(NE * 2);
  bf16* xvb   = (bf16*)alloc(NE * 2);
  bf16* xrb   = (bf16*)alloc(NE * 2);
  bf16* xgb   = (bf16*)alloc(NE * 2);
  bf16* xxxb  = (bf16*)alloc((size_t)M_DIM * 160 * 2);
  bf16* tdtmp = (bf16*)alloc((size_t)M_DIM * 64 * 2);
  // bf16 weights
  bf16* wr_b  = (bf16*)alloc(EE * 2);
  bf16* wk_b  = (bf16*)alloc(EE * 2);
  bf16* wv_b  = (bf16*)alloc(EE * 2);
  bf16* wg_b  = (bf16*)alloc(EE * 2);
  bf16* wo_b  = (bf16*)alloc(EE * 2);
  bf16* wfk_b = (bf16*)alloc(EE * 2);
  bf16* wfr_b = (bf16*)alloc(EE * 2);
  bf16* wfv_b = (bf16*)alloc(EE * 2);
  bf16* w1t   = (bf16*)alloc((size_t)160 * E_DIM * 2);     // [160, 2048]
  bf16* w2t   = (bf16*)alloc((size_t)5 * E_DIM * 32 * 2);  // 5 x [2048, 32]
  bf16* tdw1t = (bf16*)alloc((size_t)64 * E_DIM * 2);      // [64, 2048]
  bf16* tdw2t = (bf16*)alloc((size_t)E_DIM * 64 * 2);      // [2048, 64]
  // aliases (producer finished before consumer starts)
  float* xc   = xa;   // LN2 output reuses xa
  float* rrb  = sx;   // sigmoid gate reuses sx
  bf16*  oggb = xvb;  // groupnorm*gate reuses xvb
  bf16*  xk2b = xkb;
  bf16*  xr2b = xrb;
  bf16*  kkb  = z1;

  const dim3 blk(256);
  const dim3 gridEW(8192);
  const dim3 gridCast(4096);

  // ---- 0) weight precasts (bf16, [N,K] row-major == B operand layout) ----
  cast_bf16_kernel<<<gridCast, blk, 0, stream>>>(Wr,  wr_b,  EE);
  cast_bf16_kernel<<<gridCast, blk, 0, stream>>>(Wk,  wk_b,  EE);
  cast_bf16_kernel<<<gridCast, blk, 0, stream>>>(Wv,  wv_b,  EE);
  cast_bf16_kernel<<<gridCast, blk, 0, stream>>>(Wg,  wg_b,  EE);
  cast_bf16_kernel<<<gridCast, blk, 0, stream>>>(Wo,  wo_b,  EE);
  cast_bf16_kernel<<<gridCast, blk, 0, stream>>>(Wfk, wfk_b, EE);
  cast_bf16_kernel<<<gridCast, blk, 0, stream>>>(Wfr, wfr_b, EE);
  cast_bf16_kernel<<<gridCast, blk, 0, stream>>>(Wfv, wfv_b, EE);
  cast_tr_bf16_kernel<<<1024, blk, 0, stream>>>(maa_w1, w1t, E_DIM, 160);
  cast_tr_bf16_kernel<<<512,  blk, 0, stream>>>(td_w1,  tdw1t, E_DIM, 64);
  cast_tr_bf16_kernel<<<512,  blk, 0, stream>>>(td_w2,  tdw2t, 64, E_DIM);
  for (int f = 0; f < 5; ++f)
    cast_tr_bf16_kernel<<<256, blk, 0, stream>>>(maa_w2 + (size_t)f * 32 * E_DIM,
                                                 w2t + (size_t)f * E_DIM * 32, 32, E_DIM);

  // ---- 1) LN1 + token shift + maa mix ----
  ln_kernel<<<M_DIM, blk, 0, stream>>>(x, ln1_w, ln1_b, xa);
  shift_mix1_kernel<<<gridEW, blk, 0, stream>>>(xa, state, maa_x, sx, z1);

  // ---- 2) xxx = tanh(z1 @ maa_w1): M=4096, N=160, K=2048 ----
  gemm_bf16_wmma<128, 32, 8, 1, EPI_TANH_BF16>
      <<<dim3(160 / 32, M_DIM / 128), blk, 0, stream>>>(
      z1, E_DIM, w1t, E_DIM, nullptr, xxxb, 160, E_DIM, nullptr, nullptr, nullptr);

  // ---- 3) per-gate einsum + gate mix epilogue (stack order k,w,v,r,g) ----
  bf16* gates[5] = { xkb, xwb, xvb, xrb, xgb };
  for (int f = 0; f < 5; ++f)
    gemm_bf16_wmma<128, 128, 4, 2, EPI_GATE_BF16>
        <<<dim3(E_DIM / 128, M_DIM / 128), blk, 0, stream>>>(
        xxxb + f * 32, 160, w2t + (size_t)f * E_DIM * 32, 32,
        nullptr, gates[f], E_DIM, 32, xa, sx, maa_stack + (size_t)f * E_DIM);

  // ---- 4) decay chain: tanh(xw @ td_w1) @ td_w2 -> exp(-exp(td + .)) ----
  gemm_bf16_wmma<128, 64, 4, 2, EPI_TANH_BF16>
      <<<dim3(1, M_DIM / 128), blk, 0, stream>>>(
      xwb, E_DIM, tdw1t, E_DIM, nullptr, tdtmp, 64, E_DIM, nullptr, nullptr, nullptr);
  gemm_bf16_wmma<128, 128, 4, 2, EPI_DECAY_F32>
      <<<dim3(E_DIM / 128, M_DIM / 128), blk, 0, stream>>>(
      tdtmp, 64, tdw2t, 64, wb, nullptr, E_DIM, 64, nullptr, nullptr, time_decay);

  // ---- 5) r / k / v / g projections ----
  gemm_bf16_wmma<128, 128, 4, 2, EPI_F32>
      <<<dim3(E_DIM / 128, M_DIM / 128), blk, 0, stream>>>(
      xrb, E_DIM, wr_b, E_DIM, rb, nullptr, E_DIM, E_DIM, nullptr, nullptr, nullptr);
  gemm_bf16_wmma<128, 128, 4, 2, EPI_F32>
      <<<dim3(E_DIM / 128, M_DIM / 128), blk, 0, stream>>>(
      xkb, E_DIM, wk_b, E_DIM, kb2, nullptr, E_DIM, E_DIM, nullptr, nullptr, nullptr);
  gemm_bf16_wmma<128, 128, 4, 2, EPI_F32>
      <<<dim3(E_DIM / 128, M_DIM / 128), blk, 0, stream>>>(
      xvb, E_DIM, wv_b, E_DIM, vb, nullptr, E_DIM, E_DIM, nullptr, nullptr, nullptr);
  gemm_bf16_wmma<128, 128, 4, 2, EPI_SILU_F32>
      <<<dim3(E_DIM / 128, M_DIM / 128), blk, 0, stream>>>(
      xgb, E_DIM, wg_b, E_DIM, gb, nullptr, E_DIM, E_DIM, nullptr, nullptr, nullptr);

  // ---- 6) sequential WKV scan, one block per (b,h) ----
  wkv_scan_kernel<<<B_DIM * H_DIM, blk, 0, stream>>>(rb, kb2, vb, wb, u, state, ob);

  // ---- 7) groupnorm * gate, then Wo projection + residual ----
  gnorm_mul_kernel<<<M_DIM, blk, 0, stream>>>(ob, gb, gn_w, gn_b, oggb);
  gemm_bf16_wmma<128, 128, 4, 2, EPI_ADD_F32>
      <<<dim3(E_DIM / 128, M_DIM / 128), blk, 0, stream>>>(
      oggb, E_DIM, wo_b, E_DIM, x1, nullptr, E_DIM, E_DIM, x, nullptr, nullptr);

  // ---- 8) channel mixing ----
  ln_kernel<<<M_DIM, blk, 0, stream>>>(x1, ln2_w, ln2_b, xc);
  shift_mix2_kernel<<<gridEW, blk, 0, stream>>>(xc, state, ffn_maa_k, ffn_maa_r, xk2b, xr2b);
  gemm_bf16_wmma<128, 128, 4, 2, EPI_SIG_F32>
      <<<dim3(E_DIM / 128, M_DIM / 128), blk, 0, stream>>>(
      xr2b, E_DIM, wfr_b, E_DIM, rrb, nullptr, E_DIM, E_DIM, nullptr, nullptr, nullptr);
  gemm_bf16_wmma<128, 128, 4, 2, EPI_RELUSQ_BF16>
      <<<dim3(E_DIM / 128, M_DIM / 128), blk, 0, stream>>>(
      xk2b, E_DIM, wfk_b, E_DIM, nullptr, kkb, E_DIM, E_DIM, nullptr, nullptr, nullptr);
  gemm_bf16_wmma<128, 128, 4, 2, EPI_FINAL_F32>
      <<<dim3(E_DIM / 128, M_DIM / 128), blk, 0, stream>>>(
      kkb, E_DIM, wfv_b, E_DIM, (float*)d_out, nullptr, E_DIM, E_DIM, x1, rrb, nullptr);
}
